// ProteinGCN_21569325760592
// MI455X (gfx1250) — matrix operand order repaired and checked by reading, not compiled, not run on hardware
//
#include <hip/hip_runtime.h>

// ---------------------------------------------------------------------------
// ProteinGCN forward for gfx1250 (MI455X).
// Conv GEMMs run on v_wmma_f32_16x16x32_bf16; one WMMA row-tile == one atom's
// 16 neighbors, so the neighbor reduction is a column sum of the D fragment.
// Weights live in LDS pre-swizzled into B-fragment order (contiguous 32B per
// lane -> ds_load_b128 pairs); A tiles are staged as packed bf16 quads.
// ---------------------------------------------------------------------------

#define Bb    8
#define Nn    2048
#define Mm    16
#define HA    64
#define HB    32
#define KTOT  160          // 2*HA + HB
#define CO    128          // 2*HA
#define NT    8            // column tiles of 16
#define NKC   5            // K chunks of 32
#define WAVES 4            // waves per block (128 threads)
#define EPSBN 1e-5f

typedef __attribute__((ext_vector_type(16))) __bf16 v16bf;
typedef __attribute__((ext_vector_type(8)))  __bf16 v8bf;
typedef __attribute__((ext_vector_type(8)))  float  v8f;

__device__ __forceinline__ unsigned short f2bfbits(float f) {
    unsigned u = __builtin_bit_cast(unsigned, f);
    u += 0x7fffu + ((u >> 16) & 1u);              // round-to-nearest-even
    return (unsigned short)(u >> 16);
}
__device__ __forceinline__ unsigned pack2bf(float a, float b) {
    return (unsigned)f2bfbits(a) | ((unsigned)f2bfbits(b) << 16);
}

// ---------------------------------------------------------------------------
// Fused graph-conv GEMM.  APPLY==0: stats pass (per-channel sum/sumsq of z).
// APPLY==1: recompute z, BN-affine, sigmoid(filt)*relu(core), sum over the 16
// neighbor rows, emit s[B,N,64] + stats for the second BN.
// ---------------------------------------------------------------------------
template <int APPLY>
__global__ __launch_bounds__(128) void conv_wmma(
    const float* __restrict__ emb,   // [B*N,64]  (pre-masked)
    const int*   __restrict__ adj,   // [B*N,16]
    const float* __restrict__ nbr,   // [B*N,16,32]
    const float* __restrict__ mask,  // [B*N]
    const float* __restrict__ W,     // [160,128] layer slice
    const float* __restrict__ bias,  // [128]
    const float* __restrict__ bnsc,  // [128] BN1 scale   (APPLY only)
    const float* __restrict__ bnsh,  // [128] BN1 shift   (APPLY only)
    float*       __restrict__ gstat, // [256] sum|sumsq   (stats pass)
    float*       __restrict__ sout,  // [B*N,64]          (APPLY only)
    float*       __restrict__ gstat2)// [128] sum|sumsq   (APPLY only)
{
    // Weights in B-fragment order: index ((kc*NT+ct)*32 + lane)*16 + j
    __shared__ __align__(32) unsigned short sW[NKC * NT * 32 * 16]; // 40 KB
    __shared__ __align__(32) unsigned short sA[WAVES][16 * KTOT];   // 20 KB
    __shared__ float sStat[2 * CO];                                 // 1 KB

    const int tid  = threadIdx.x;
    const int w    = tid >> 5;
    const int lane = tid & 31;

    for (int i = tid; i < 2 * CO; i += blockDim.x) sStat[i] = 0.f;
    // fill weights pre-swizzled into fragment order
    for (int i = tid; i < NKC * NT * 32 * 16; i += blockDim.x) {
        const int j  = i & 15;
        const int ln = (i >> 4) & 31;
        const int ct = (i >> 9) & 7;
        const int kc = i >> 12;
        const int k  = kc * 32 + ((ln >> 4) << 3) + (j & 7) + ((j >> 3) << 4);   // B K: 8h+{0..7}, 16+8h+{0..7}
        const int c  = ct * 16 + (ln & 15);
        sW[i] = f2bfbits(W[k * CO + c]);
    }
    __syncthreads();

    const int wstride = gridDim.x * WAVES;
    for (int at = blockIdx.x * WAVES + w; at < Bb * Nn; at += wstride) {
        const int   b      = at >> 11;        // at / 2048
        const float mk     = mask[at];
        const long  base64 = (long)at * HA;
        const float* selfp = emb + base64;

        // ---- stage A tile (16 rows x 160) into LDS as packed bf16 ----------
        for (int r = 0; r < 16; ++r) {
            const int nbi = adj[at * Mm + r];
            const float* nbrp  = emb + (long)(b * Nn + nbi) * HA;
            const float* bondp = nbr + ((long)at * Mm + r) * HB;
            for (int cq = lane; cq < KTOT / 4; cq += 32) {   // 40 quads
                const int c = cq * 4;                         // never straddles segs
                float4 v;
                if (c < HA) {
                    v = *(const float4*)(selfp + c);
                } else if (c < 2 * HA) {
                    v = *(const float4*)(nbrp + (c - HA));
                    v.x *= mk; v.y *= mk; v.z *= mk; v.w *= mk;
                } else {
                    v = *(const float4*)(bondp + (c - 2 * HA));
                }
                uint2 p;
                p.x = pack2bf(v.x, v.y);
                p.y = pack2bf(v.z, v.w);
                *(uint2*)&sA[w][r * KTOT + c] = p;           // ds_store_b64
            }
        }
        // DS ops are in-order per wave; waiting on this wave's dscnt makes the
        // whole staged tile visible to all 32 lanes (lockstep wave32).
        asm volatile("s_wait_dscnt 0" ::: "memory");

        // ---- A fragments per documented 16-bit 16x32 layout ---------------
        // lane holds row (lane&15); K = koff+{0..7} and koff+{16..23},
        // koff = 8*(lane>>4): two contiguous 16B runs -> two ds_load_b128.
        v16bf afrag[NKC];
        {
            const int arow = lane & 15;
            const int koff = (lane >> 4) << 3;
            const unsigned short* ap0 = &sA[w][arow * KTOT];
            #pragma unroll
            for (int kc = 0; kc < NKC; ++kc) {
                const unsigned short* ap = ap0 + kc * 32 + koff;
                v8bf lo = *(const v8bf*)(ap);
                v8bf hi = *(const v8bf*)(ap + 16);
                #pragma unroll
                for (int j = 0; j < 8; ++j) {
                    afrag[kc][j]     = lo[j];
                    afrag[kc][j + 8] = hi[j];
                }
            }
        }

        v8f acc[NT] = {};
        const int coln = lane & 15;

        #pragma unroll
        for (int ct = 0; ct < NT; ++ct) {
            #pragma unroll
            for (int kc = 0; kc < NKC; ++kc) {
                // contiguous 32B per lane -> single v16bf LDS load
                const v16bf bfrag =
                    *(const v16bf*)&sW[(((kc * NT + ct) * 32) + lane) * 16];
                acc[ct] = __builtin_amdgcn_wmma_f32_16x16x32_bf16(
                    false, afrag[kc], false, bfrag, (short)0, acc[ct],
                    false, false);
            }
        }

        if (APPLY == 0) {
            // per-channel partial sum / sumsq of z = acc + bias
            #pragma unroll
            for (int ct = 0; ct < NT; ++ct) {
                const int   col = ct * 16 + coln;
                const float bi  = bias[col];
                float ls = 0.f, lq = 0.f;
                #pragma unroll
                for (int j = 0; j < 8; ++j) {
                    const float z = acc[ct][j] + bi;
                    ls += z; lq += z * z;
                }
                atomicAdd(&sStat[col], ls);
                atomicAdd(&sStat[CO + col], lq);
            }
        } else {
            // BN1-affine + gate + neighbor-sum; tiles 0..3 = filt, 4..7 = core
            #pragma unroll
            for (int t = 0; t < 4; ++t) {
                const int   cf  = t * 16 + coln;
                const int   cc  = cf + HA;
                const float scf = bnsc[cf], shf = bnsh[cf] + bias[cf] * scf;
                const float scc = bnsc[cc], shc = bnsh[cc] + bias[cc] * scc;
                float part = 0.f;
                #pragma unroll
                for (int j = 0; j < 8; ++j) {
                    const float zf = acc[t][j]     * scf + shf;
                    const float zc = acc[t + 4][j] * scc + shc;
                    part += (1.f / (1.f + __expf(-zf))) * fmaxf(zc, 0.f);
                }
                part += __shfl_xor(part, 16, 32);  // combine row halves
                if (lane < 16) {
                    sout[base64 + cf] = part;
                    atomicAdd(&sStat[cf], part);
                    atomicAdd(&sStat[HA + cf], part * part);
                }
            }
        }
    }
    __syncthreads();
    if (APPLY == 0) {
        for (int i = tid; i < 2 * CO; i += blockDim.x) atomicAdd(&gstat[i], sStat[i]);
    } else {
        for (int i = tid; i < 2 * HA; i += blockDim.x) atomicAdd(&gstat2[i], sStat[i]);
    }
}

// ---------------------------------------------------------------------------
// Small helper kernels
// ---------------------------------------------------------------------------
__global__ void fill_f32(float* p, float v, int n) {
    int i = blockIdx.x * blockDim.x + threadIdx.x;
    if (i < n) p[i] = v;
}
__global__ void fill_i32(int* p, int v, int n) {
    int i = blockIdx.x * blockDim.x + threadIdx.x;
    if (i < n) p[i] = v;
}

__global__ void bn_finalize(const float* __restrict__ stat,
                            const float* __restrict__ g,
                            const float* __restrict__ be,
                            float cnt, int nch,
                            float* __restrict__ scale,
                            float* __restrict__ shift) {
    int c = blockIdx.x * blockDim.x + threadIdx.x;
    if (c < nch) {
        const float m  = stat[c] / cnt;
        const float v  = stat[nch + c] / cnt - m * m;
        const float sc = g[c] * rsqrtf(v + EPSBN);
        scale[c] = sc;
        shift[c] = be[c] - m * sc;
    }
}

__global__ void embed_k(const int* __restrict__ idx,
                        const float* __restrict__ table,  // [100,92]
                        const float* __restrict__ W,      // [92,64]
                        const float* __restrict__ bv,     // [64]
                        const float* __restrict__ mask,
                        float* __restrict__ out) {        // [B*N,64] masked
    const int bn = blockIdx.x;
    const int h  = threadIdx.x;
    const float* row = table + (long)idx[bn] * 92;
    float a = bv[h];
    for (int k = 0; k < 92; ++k) a += row[k] * W[k * HA + h];
    out[(long)bn * HA + h] = a * mask[bn];
}

__global__ void residual_update(const float* __restrict__ ein,
                                const float* __restrict__ s,
                                const float* __restrict__ sc,
                                const float* __restrict__ sh,
                                const float* __restrict__ mask,
                                float* __restrict__ eout, int total) {
    int i = blockIdx.x * blockDim.x + threadIdx.x;
    if (i < total) {
        const int h = i & (HA - 1);
        float v = ein[i] + s[i] * sc[h] + sh[h];
        v = fmaxf(v, 0.f);
        eout[i] = v * mask[i >> 6];   // pre-mask for next layer / pooling
    }
}

__global__ void pool_scatter(const float* __restrict__ emb,
                             const int* __restrict__ aidx,
                             const float* __restrict__ mask,
                             float* __restrict__ amino,    // [2048,64]
                             float* __restrict__ prot,     // [8,64]
                             float* __restrict__ msum,     // [8]
                             int* __restrict__ gmin) {
    const int bn = blockIdx.x;
    const int h  = threadIdx.x;
    const float v = emb[(long)bn * HA + h];
    atomicAdd(&amino[(long)aidx[bn] * HA + h], v);
    atomicAdd(&prot[(long)(bn >> 11) * HA + h], v);
    if (h == 0) {
        atomicAdd(&msum[bn >> 11], mask[bn]);
        atomicMin(gmin, aidx[bn]);
    }
}

__global__ void amino_head(const float* __restrict__ amino,
                           const float* __restrict__ Waf,   // [64,32]
                           const float* __restrict__ baf,   // [32]
                           const float* __restrict__ Wao,   // [32,1]
                           const float* __restrict__ bao,   // [1]
                           float* __restrict__ out) {
    const int a = blockIdx.x;
    const int g = threadIdx.x;                 // 32 threads = 1 wave
    float t = baf[g];
    for (int h = 0; h < HA; ++h)
        t += fmaxf(amino[(long)a * HA + h], 0.f) * Waf[h * 32 + g];
    t = fmaxf(t, 0.f);
    float v = t * Wao[g];
    for (int o = 16; o > 0; o >>= 1) v += __shfl_xor(v, o, 32);
    if (g == 0) out[Bb + a] = v + bao[0];
}

__global__ void protein_head(const float* __restrict__ prot,
                             const float* __restrict__ msum,
                             const float* __restrict__ Wcf,  // [64,32]
                             const float* __restrict__ bcf,  // [32]
                             const float* __restrict__ Wout, // [32,1]
                             const float* __restrict__ bout, // [1]
                             float* __restrict__ out) {
    const int b = blockIdx.x;
    const int g = threadIdx.x;
    const float inv = 1.f / msum[b];
    float t = bcf[g];
    for (int h = 0; h < HA; ++h)
        t += fmaxf(prot[(long)b * HA + h] * inv, 0.f) * Wcf[h * 32 + g];
    t = fmaxf(t, 0.f);
    float v = t * Wout[g];
    for (int o = 16; o > 0; o >>= 1) v += __shfl_xor(v, o, 32);
    if (g == 0) out[b] = v + bout[0];
}

__global__ void mask_pooled_k(const int* __restrict__ gmin, float* __restrict__ out) {
    int a = blockIdx.x * blockDim.x + threadIdx.x;
    if (a < Nn) out[Bb + Nn + a] = (a < *gmin) ? 0.f : 1.f;
}

// ---------------------------------------------------------------------------
// Workspace layout (float offsets)
// ---------------------------------------------------------------------------
#define O_EMB0  0
#define O_EMB1  (Bb * Nn * HA)                 // 1048576
#define O_S     (2 * Bb * Nn * HA)
#define O_STAT1 (3 * Bb * Nn * HA)             // 256
#define O_STAT2 (O_STAT1 + 256)                // 128
#define O_BN1SC (O_STAT2 + 128)                // 128
#define O_BN1SH (O_BN1SC + 128)                // 128
#define O_BN2SC (O_BN1SH + 128)                // 64
#define O_BN2SH (O_BN2SC + 64)                 // 64
#define O_AMINO (O_BN2SH + 64)                 // 2048*64
#define O_PROT  (O_AMINO + Nn * HA)            // 512
#define O_MSUM  (O_PROT + Bb * HA)             // 8
#define O_GMIN  (O_MSUM + Bb)                  // 1 (int)

extern "C" void kernel_launch(void* const* d_in, const int* in_sizes, int n_in,
                              void* d_out, int out_size, void* d_ws, size_t ws_size,
                              hipStream_t stream) {
    const int*   atom_idx  = (const int*)  d_in[0];
    const float* nbr       = (const float*)d_in[1];
    const int*   adj       = (const int*)  d_in[2];
    const int*   amino_idx = (const int*)  d_in[3];
    const float* amask     = (const float*)d_in[4];
    const float* table     = (const float*)d_in[5];
    const float* W_emb     = (const float*)d_in[6];
    const float* b_emb     = (const float*)d_in[7];
    const float* fcW       = (const float*)d_in[8];
    const float* fcb       = (const float*)d_in[9];
    const float* bnh_g     = (const float*)d_in[10];
    const float* bnh_b     = (const float*)d_in[11];
    const float* bno_g     = (const float*)d_in[12];
    const float* bno_b     = (const float*)d_in[13];
    const float* W_cf      = (const float*)d_in[14];
    const float* b_cf      = (const float*)d_in[15];
    const float* W_out     = (const float*)d_in[16];
    const float* b_out     = (const float*)d_in[17];
    const float* W_af      = (const float*)d_in[18];
    const float* b_af      = (const float*)d_in[19];
    const float* W_ao      = (const float*)d_in[20];
    const float* b_ao      = (const float*)d_in[21];

    float* out = (float*)d_out;
    float* ws  = (float*)d_ws;

    float* emb0  = ws + O_EMB0;
    float* emb1  = ws + O_EMB1;
    float* sbuf  = ws + O_S;
    float* stat1 = ws + O_STAT1;
    float* stat2 = ws + O_STAT2;
    float* bn1sc = ws + O_BN1SC;
    float* bn1sh = ws + O_BN1SH;
    float* bn2sc = ws + O_BN2SC;
    float* bn2sh = ws + O_BN2SH;
    float* amino = ws + O_AMINO;
    float* prot  = ws + O_PROT;
    float* msum  = ws + O_MSUM;
    int*   gmin  = (int*)(ws + O_GMIN);

    // initial embedding GEMM (small, VALU) -> masked atom_emb
    embed_k<<<Bb * Nn, HA, 0, stream>>>(atom_idx, table, W_emb, b_emb, amask, emb0);

    float* cur = emb0;
    float* nxt = emb1;
    for (int layer = 0; layer < 4; ++layer) {
        const float* Wl = fcW + (long)layer * KTOT * CO;
        const float* bl = fcb + (long)layer * CO;

        fill_f32<<<2, 256, 0, stream>>>(stat1, 0.f, 384);  // stat1(256)+stat2(128)

        conv_wmma<0><<<1024, 128, 0, stream>>>(cur, adj, nbr, amask, Wl, bl,
                                               nullptr, nullptr, stat1,
                                               nullptr, nullptr);
        bn_finalize<<<1, 128, 0, stream>>>(stat1, bnh_g + layer * CO,
                                           bnh_b + layer * CO,
                                           (float)(Bb * Nn * Mm), CO,
                                           bn1sc, bn1sh);
        conv_wmma<1><<<1024, 128, 0, stream>>>(cur, adj, nbr, amask, Wl, bl,
                                               bn1sc, bn1sh, nullptr,
                                               sbuf, stat2);
        bn_finalize<<<1, 64, 0, stream>>>(stat2, bno_g + layer * HA,
                                          bno_b + layer * HA,
                                          (float)(Bb * Nn), HA,
                                          bn2sc, bn2sh);
        residual_update<<<(Bb * Nn * HA + 255) / 256, 256, 0, stream>>>(
            cur, sbuf, bn2sc, bn2sh, amask, nxt, Bb * Nn * HA);

        float* t = cur; cur = nxt; nxt = t;
    }

    // pooling
    const int poolN = Nn * HA + Bb * HA + Bb;  // amino + prot + msum (contiguous)
    fill_f32<<<(poolN + 255) / 256, 256, 0, stream>>>(amino, 0.f, poolN);
    fill_i32<<<1, 32, 0, stream>>>(gmin, 0x7fffffff, 1);

    pool_scatter<<<Bb * Nn, HA, 0, stream>>>(cur, amino_idx, amask,
                                             amino, prot, msum, gmin);

    amino_head<<<Nn, 32, 0, stream>>>(amino, W_af, b_af, W_ao, b_ao, out);
    protein_head<<<Bb, 32, 0, stream>>>(prot, msum, W_cf, b_cf, W_out, b_out, out);
    mask_pooled_k<<<Nn / 256, 256, 0, stream>>>(gmin, out);
}
